// RNNForParityCheck_51608327029372
// MI455X (gfx1250) — matrix operand order repaired and checked by reading, not compiled, not run on hardware
//
#include <hip/hip_runtime.h>
#include <hip/hip_bf16.h>
#include <math.h>

typedef _Float16 v8h  __attribute__((ext_vector_type(8)));
typedef _Float16 v16h __attribute__((ext_vector_type(16)));
typedef float    v8f  __attribute__((ext_vector_type(8)));

#define Bn   128     // batch
#define Tn   256     // time steps
#define Cn   512     // hidden
#define NBLK 32      // persistent workgroups
#define NWG  16      // N columns per workgroup (Cn / NBLK)
#define KPAD 528     // padded K stride in LDS (halves): 1056B rows, 32B aligned

// ---------------------------------------------------------------------------
// Grid-wide barrier (sense-reversing, agent scope). bar[0]=count, bar[1]=gen.
// Reset each launch by prep kernel, so graph replay is deterministic.
// ---------------------------------------------------------------------------
__device__ __forceinline__ void grid_sync(unsigned* bar, unsigned nblocks) {
    __syncthreads();
    if (threadIdx.x == 0) {
        unsigned* cnt = bar;
        unsigned* gen = bar + 1;
        unsigned g = __hip_atomic_load(gen, __ATOMIC_RELAXED, __HIP_MEMORY_SCOPE_AGENT);
        if (__hip_atomic_fetch_add(cnt, 1u, __ATOMIC_ACQ_REL, __HIP_MEMORY_SCOPE_AGENT)
                == nblocks - 1u) {
            __hip_atomic_store(cnt, 0u, __ATOMIC_RELAXED, __HIP_MEMORY_SCOPE_AGENT);
            __hip_atomic_store(gen, g + 1u, __ATOMIC_RELEASE, __HIP_MEMORY_SCOPE_AGENT);
        } else {
            while (__hip_atomic_load(gen, __ATOMIC_ACQUIRE, __HIP_MEMORY_SCOPE_AGENT) == g) {
                __builtin_amdgcn_s_sleep(1);
            }
        }
    }
    __syncthreads();
}

// One 16x16x32 f16 WMMA step: A from global hidden-state buffer, B from LDS.
// aptr already includes the lane's (row, koffA) offset; bptr includes (n, koffB).
__device__ __forceinline__ v8f wmma_step(const _Float16* __restrict__ aptr,
                                         const _Float16* bptr, v8f acc) {
    v16h a = *(const v16h*)(aptr);       // K 0..15 (lanes<16) / 16..31 (lanes>=16)
    v16h b = *(const v16h*)(bptr);       // K 0..15 (lanes<16) / 16..31 (lanes>=16)
    return __builtin_amdgcn_wmma_f32_16x16x32_f16(false, a, false, b,
                                                  (short)0, acc, false, false);
}

// ---------------------------------------------------------------------------
// Prep: P0 table, fused layer-1 bias, f16 weight conversion, zero hidden
// state, reset barrier.
// ---------------------------------------------------------------------------
__global__ void rnn_prep_kernel(const float* __restrict__ W_emb,   // [C,V]
                                const float* __restrict__ b_emb,   // [C]
                                const float* __restrict__ W_ih,    // [L,C,C]
                                const float* __restrict__ b_ih,    // [L,C]
                                const float* __restrict__ W_hh,    // [L,C,C]
                                const float* __restrict__ b_hh,    // [L,C]
                                float* __restrict__ P0,            // [2,C]
                                float* __restrict__ bias1,         // [C]
                                unsigned* __restrict__ bar,
                                _Float16* __restrict__ Whh0h,
                                _Float16* __restrict__ Wih1h,
                                _Float16* __restrict__ Whh1h,
                                _Float16* __restrict__ h0,         // [2,B,C]
                                _Float16* __restrict__ h1)         // [2,B,C]
{
    const int gtid = blockIdx.x * blockDim.x + threadIdx.x;
    const int nth  = gridDim.x * blockDim.x;

    for (int i = gtid; i < Cn * Cn; i += nth) {
        Whh0h[i] = (_Float16)W_hh[i];
        Wih1h[i] = (_Float16)W_ih[Cn * Cn + i];
        Whh1h[i] = (_Float16)W_hh[Cn * Cn + i];
    }
    for (int i = gtid; i < 2 * Bn * Cn; i += nth) {
        h0[i] = (_Float16)0.f;
        h1[i] = (_Float16)0.f;
    }
    if (gtid < 2 * Cn) {          // P0[v][c]: embedding row folded through W_ih0
        const int c = gtid >> 1, v = gtid & 1;
        float s = 0.f;
        for (int k = 0; k < Cn; ++k)
            s += (W_emb[k * 2 + v] + b_emb[k]) * W_ih[c * Cn + k];
        P0[v * Cn + c] = s + b_ih[c] + b_hh[c];
    }
    if (gtid < Cn) bias1[gtid] = b_ih[Cn + gtid] + b_hh[Cn + gtid];
    if (gtid == 0) { bar[0] = 0u; bar[1] = 0u; }
}

// ---------------------------------------------------------------------------
// Persistent RNN: 32 blocks x 256 threads (8 wave32). Block owns 16 N-cols,
// wave owns 16 M-rows. Weights staged in LDS once; 1 grid barrier per step.
// ---------------------------------------------------------------------------
__global__ void __launch_bounds__(256, 1)
rnn_persistent_kernel(const int* __restrict__ ids,       // [B,T]
                      const float* __restrict__ P0,      // [2,C]
                      const float* __restrict__ bias1,   // [C]
                      const _Float16* __restrict__ Whh0h,
                      const _Float16* __restrict__ Wih1h,
                      const _Float16* __restrict__ Whh1h,
                      _Float16* __restrict__ h0,         // [2,B,C] ping-pong
                      _Float16* __restrict__ h1,         // [2,B,C] ping-pong
                      unsigned* __restrict__ bar,
                      float* __restrict__ S_out)         // d_out + 256 : [2,B,C]
{
    extern __shared__ _Float16 lds[];   // [3][NWG][KPAD]
    const int tid   = threadIdx.x;
    const int wv    = tid >> 5;         // 0..7  -> M-tile
    const int lane  = tid & 31;
    const int nbase = blockIdx.x * NWG;

    // Stage this block's N-slice of the three weight matrices into LDS.
    for (int idx = tid; idx < 3 * NWG * (Cn / 2); idx += 256) {
        const int which = idx / (NWG * (Cn / 2));
        const int rem   = idx - which * (NWG * (Cn / 2));
        const int nl    = rem / (Cn / 2);
        const int kw    = rem - nl * (Cn / 2);
        const _Float16* src =
            (which == 0 ? Whh0h : which == 1 ? Wih1h : Whh1h) + (nbase + nl) * Cn;
        ((uint32_t*)(lds + (which * NWG + nl) * KPAD))[kw] = ((const uint32_t*)src)[kw];
    }
    __syncthreads();

    const int r     = lane & 15;
    const int koffA = (lane < 16) ? 0 : 16;   // A-matrix contiguous K layout (matches B)
    const int koffB = (lane < 16) ? 0 : 16;   // B-matrix contiguous K layout (matches A)
    const int arow  = wv * 16 + r;            // batch row this lane reads for A
    const int n     = nbase + r;              // output column this lane owns
    const _Float16* bp0 = lds + (0 * NWG + r) * KPAD + koffB;   // W_hh0 slice
    const _Float16* bp1 = lds + (1 * NWG + r) * KPAD + koffB;   // W_ih1 slice
    const _Float16* bp2 = lds + (2 * NWG + r) * KPAD + koffB;   // W_hh1 slice
    const float bb1 = bias1[n];

    for (int t = 0; t < Tn; ++t) {
        const int cur = t & 1, prv = cur ^ 1;
        const bool last = (t == Tn - 1);

        // ---- layer 0: h0_new = tanh(h0_old @ Whh0^T + P0[id]) ----
        {
            const _Float16* aptr = h0 + prv * Bn * Cn + arow * Cn + koffA;
            v8f acc = {};
            #pragma unroll
            for (int kk = 0; kk < Cn; kk += 32)
                acc = wmma_step(aptr + kk, bp0 + kk, acc);

            _Float16* hc = h0 + cur * Bn * Cn;
            #pragma unroll
            for (int j = 0; j < 8; ++j) {   // C/D layout: VGPR j -> M = j (+8 hi lanes)
                const int m  = wv * 16 + j + ((lane >> 4) << 3);
                const int id = ids[m * Tn + t];
                const float y = tanhf(acc[j] + P0[id * Cn + n]);
                hc[m * Cn + n] = (_Float16)y;
                if (last) S_out[m * Cn + n] = y;             // S[0]
            }
        }

        grid_sync(bar, NBLK);   // all h0_new visible before layer 1 consumes it

        // ---- layer 1: h1_new = tanh(h0_new @ Wih1^T + h1_old @ Whh1^T + b) ----
        {
            const _Float16* aptr0 = h0 + cur * Bn * Cn + arow * Cn + koffA;
            const _Float16* aptr1 = h1 + prv * Bn * Cn + arow * Cn + koffA;
            v8f acc = {};
            #pragma unroll
            for (int kk = 0; kk < Cn; kk += 32)
                acc = wmma_step(aptr0 + kk, bp1 + kk, acc);
            #pragma unroll
            for (int kk = 0; kk < Cn; kk += 32)
                acc = wmma_step(aptr1 + kk, bp2 + kk, acc);

            _Float16* hc = h1 + cur * Bn * Cn;
            #pragma unroll
            for (int j = 0; j < 8; ++j) {
                const int m = wv * 16 + j + ((lane >> 4) << 3);
                const float y = tanhf(acc[j] + bb1);
                hc[m * Cn + n] = (_Float16)y;
                if (last) S_out[Bn * Cn + m * Cn + n] = y;   // S[1]
            }
        }
        // next iteration's grid_sync covers the remaining WAR hazards
        // (ping-pong buffers make one barrier per step sufficient).
    }
}

// ---------------------------------------------------------------------------
// Head: logits = S1 @ W_cls^T + b_cls; log_softmax; NLL loss. One block.
// ---------------------------------------------------------------------------
__global__ void rnn_head_kernel(const float* __restrict__ W_cls,  // [2,C]
                                const float* __restrict__ b_cls,  // [2]
                                const int*  __restrict__ labels,  // [B]
                                float* __restrict__ out)          // d_out
{
    const float* S1 = out + 256 + Bn * Cn;   // S[1] written by rnn kernel
    __shared__ float lg[2 * Bn];
    __shared__ float lp[Bn];
    const int tid = threadIdx.x;
    const int b = tid >> 1, l = tid & 1;

    float s = b_cls[l];
    for (int c = 0; c < Cn; ++c) s += S1[b * Cn + c] * W_cls[l * Cn + c];
    out[b * 2 + l] = s;                       // logits
    lg[tid] = s;
    __syncthreads();

    if (tid < Bn) {
        const float l0 = lg[2 * tid], l1 = lg[2 * tid + 1];
        const float mx = fmaxf(l0, l1);
        const float lse = mx + logf(expf(l0 - mx) + expf(l1 - mx));
        lp[tid] = (labels[tid] == 0 ? l0 : l1) - lse;
    }
    __syncthreads();
    if (tid == 0) {
        float acc = 0.f;
        for (int i = 0; i < Bn; ++i) acc += lp[i];
        out[256 + 2 * Bn * Cn] = -acc / (float)Bn;   // loss
    }
}

// ---------------------------------------------------------------------------
extern "C" void kernel_launch(void* const* d_in, const int* in_sizes, int n_in,
                              void* d_out, int out_size, void* d_ws, size_t ws_size,
                              hipStream_t stream) {
    const int*   ids    = (const int*)  d_in[0];
    const int*   labels = (const int*)  d_in[1];
    const float* W_emb  = (const float*)d_in[2];
    const float* b_emb  = (const float*)d_in[3];
    const float* W_ih   = (const float*)d_in[4];
    const float* b_ih   = (const float*)d_in[5];
    const float* W_hh   = (const float*)d_in[6];
    const float* b_hh   = (const float*)d_in[7];
    const float* W_cls  = (const float*)d_in[8];
    const float* b_cls  = (const float*)d_in[9];
    float* out = (float*)d_out;

    // Workspace layout (all chunks 16B-aligned): ~2.0 MB total
    float*    P0    = (float*)d_ws;                    // 1024 f32
    float*    bias1 = P0 + 1024;                       //  512 f32
    unsigned* bar   = (unsigned*)(bias1 + 512);        //   8 u32 slot
    _Float16* wh    = (_Float16*)((char*)bar + 32);
    _Float16* Whh0h = wh;                              // 512KB
    _Float16* Wih1h = Whh0h + Cn * Cn;                 // 512KB
    _Float16* Whh1h = Wih1h + Cn * Cn;                 // 512KB
    _Float16* h0    = Whh1h + Cn * Cn;                 // 2*B*C f16 (ping-pong)
    _Float16* h1    = h0 + 2 * Bn * Cn;

    rnn_prep_kernel<<<64, 256, 0, stream>>>(W_emb, b_emb, W_ih, b_ih, W_hh, b_hh,
                                            P0, bias1, bar, Whh0h, Wih1h, Whh1h,
                                            h0, h1);

    const size_t ldsBytes = (size_t)3 * NWG * KPAD * sizeof(_Float16);  // ~49.5KB
    rnn_persistent_kernel<<<NBLK, 256, ldsBytes, stream>>>(
        ids, P0, bias1, Whh0h, Wih1h, Whh1h, h0, h1, bar, out + 256);

    rnn_head_kernel<<<1, 256, 0, stream>>>(W_cls, b_cls, labels, out);
}
